// GNN_60790967108277
// MI455X (gfx1250) — matrix operand order repaired and checked by reading, not compiled, not run on hardware
//
#include <hip/hip_runtime.h>
#include <hip/hip_bf16.h>

#define NNODES 100000
#define NEDGES 1600000
#define DIM    128
#define DIM2   256
#define NLAYERS 2
#define BN_EPS 1e-5f

typedef __attribute__((ext_vector_type(16))) __bf16 bf16x16;
typedef __attribute__((ext_vector_type(8)))  float  f32x8;

// ---------------------------------------------------------------------------
// h0[n,:] = sum_f atom_emb[f, x[n,f], :] + z_emb[z[n], :]
// one wave32 per node, lane handles 4 channels (float4)
// ---------------------------------------------------------------------------
__global__ void __launch_bounds__(256)
gin_node_embed(const int* __restrict__ x, const int* __restrict__ z,
               const float* __restrict__ atom_emb, const float* __restrict__ z_emb,
               float* __restrict__ h) {
  int wave = (blockIdx.x * blockDim.x + threadIdx.x) >> 5;
  int lane = threadIdx.x & 31;
  if (wave >= NNODES) return;
  int c = lane * 4;
  float4 acc = *(const float4*)(z_emb + (size_t)z[wave] * DIM + c);
#pragma unroll
  for (int f = 0; f < 9; ++f) {
    int idx = x[wave * 9 + f];
    float4 t = *(const float4*)(atom_emb + ((size_t)f * 119 + idx) * DIM + c);
    acc.x += t.x; acc.y += t.y; acc.z += t.z; acc.w += t.w;
  }
  *(float4*)(h + (size_t)wave * DIM + c) = acc;
}

// ---------------------------------------------------------------------------
// one wave32 per edge: msg = relu(h[src] + bond_emb) ; atomic scatter-add
// h/agg fit in 192MB L2 -> atomics stay on-chip
// ---------------------------------------------------------------------------
__global__ void __launch_bounds__(256)
gin_edge_message(const int* __restrict__ edge_index, const int* __restrict__ edge_attr,
                 const float* __restrict__ bond_tab, // [3,6,128] for this layer
                 const float* __restrict__ h, float* __restrict__ agg) {
  int wave = (blockIdx.x * blockDim.x + threadIdx.x) >> 5;
  int lane = threadIdx.x & 31;
  if (wave >= NEDGES) return;
  int src = edge_index[wave];
  int dst = edge_index[NEDGES + wave];
  int a0 = edge_attr[wave * 3 + 0];
  int a1 = edge_attr[wave * 3 + 1];
  int a2 = edge_attr[wave * 3 + 2];
  int c = lane * 4;
  float4 m  = *(const float4*)(h + (size_t)src * DIM + c);
  float4 t0 = *(const float4*)(bond_tab + (0 * 6 + a0) * DIM + c);
  float4 t1 = *(const float4*)(bond_tab + (1 * 6 + a1) * DIM + c);
  float4 t2 = *(const float4*)(bond_tab + (2 * 6 + a2) * DIM + c);
  float v0 = fmaxf(m.x + t0.x + t1.x + t2.x, 0.f);
  float v1 = fmaxf(m.y + t0.y + t1.y + t2.y, 0.f);
  float v2 = fmaxf(m.z + t0.z + t1.z + t2.z, 0.f);
  float v3 = fmaxf(m.w + t0.w + t1.w + t2.w, 0.f);
  float* out = agg + (size_t)dst * DIM + c;
  unsafeAtomicAdd(out + 0, v0);
  unsafeAtomicAdd(out + 1, v1);
  unsafeAtomicAdd(out + 2, v2);
  unsafeAtomicAdd(out + 3, v3);
}

// ---------------------------------------------------------------------------
// pre-swizzle W[K,NOUT] (f32 row-major) into per-lane WMMA B fragments (bf16):
// fragment element t of lane L of tile (ktile,ntile):
//   col = ntile*16 + (L&15), K = ktile*32 + (L>>4)*8 + (t&7) + (t>>3)*16
// stored contiguously so the GEMM does one coalesced 32B load per lane.
// ---------------------------------------------------------------------------
__global__ void gin_convert_w(const float* __restrict__ W, __bf16* __restrict__ out,
                              int K, int NOUT) {
  int tid = blockIdx.x * blockDim.x + threadIdx.x;
  if (tid >= K * NOUT) return;
  int t     = tid & 15;
  int lane  = (tid >> 4) & 31;
  int rest  = tid >> 9;
  int ntiles = NOUT >> 4;
  int ntile = rest % ntiles;
  int ktile = rest / ntiles;
  int k = ktile * 32 + (lane >> 4) * 8 + (t & 7) + (t >> 3) * 16;
  int n = ntile * 16 + (lane & 15);
  out[tid] = (__bf16)W[(size_t)k * NOUT + n];
}

// ---------------------------------------------------------------------------
// GEMM1: C1[N,256] = ((1+eps)*h + agg) @ W1   (A built in-register, bf16)
// one wave per 16x16 tile, K=128 -> 4 x v_wmma_f32_16x16x32_bf16
// ---------------------------------------------------------------------------
__global__ void __launch_bounds__(256)
gin_gemm1(const float* __restrict__ h, const float* __restrict__ agg,
          const float* __restrict__ epsv, int layer,
          const __bf16* __restrict__ Bswz, float* __restrict__ C) {
  const int NT = DIM2 / 16; // 16 col tiles
  int wave = (blockIdx.x * blockDim.x + threadIdx.x) >> 5;
  int lane = threadIdx.x & 31;
  int mtile = wave / NT;
  int ntile = wave % NT;
  if (mtile >= NNODES / 16) return;
  float coef = 1.0f + epsv[layer];
  int row = mtile * 16 + (lane & 15);
  int kh  = lane >> 4;
  f32x8 acc = {0.f, 0.f, 0.f, 0.f, 0.f, 0.f, 0.f, 0.f};
#pragma unroll
  for (int kk = 0; kk < DIM; kk += 32) {
    const float* hp = h   + (size_t)row * DIM + kk + kh * 8;
    const float* gp = agg + (size_t)row * DIM + kk + kh * 8;
    bf16x16 a;
#pragma unroll
    for (int j = 0; j < 8; ++j) a[j]     = (__bf16)(coef * hp[j]      + gp[j]);
#pragma unroll
    for (int j = 0; j < 8; ++j) a[8 + j] = (__bf16)(coef * hp[16 + j] + gp[16 + j]);
    bf16x16 b = *(const bf16x16*)(Bswz + (((size_t)(kk >> 5) * NT + ntile) * 32 + lane) * 16);
    acc = __builtin_amdgcn_wmma_f32_16x16x32_bf16(false, a, false, b, (short)0, acc, false, false);
  }
  int col   = ntile * 16 + (lane & 15);
  int rbase = mtile * 16 + kh * 8;
#pragma unroll
  for (int i = 0; i < 8; ++i) C[(size_t)(rbase + i) * DIM2 + col] = acc[i];
}

// ---------------------------------------------------------------------------
// GEMM2: C2[N,128] = relu(BN1(C1)) @ W2 ; BN affine fused into A-fragment load
// K=256 -> 8 x v_wmma_f32_16x16x32_bf16 per tile
// ---------------------------------------------------------------------------
__global__ void __launch_bounds__(256)
gin_gemm2(const float* __restrict__ C1, const float* __restrict__ scale,
          const float* __restrict__ shift,
          const __bf16* __restrict__ Bswz, float* __restrict__ C2) {
  const int NT = DIM / 16; // 8 col tiles
  int wave = (blockIdx.x * blockDim.x + threadIdx.x) >> 5;
  int lane = threadIdx.x & 31;
  int mtile = wave / NT;
  int ntile = wave % NT;
  if (mtile >= NNODES / 16) return;
  int row = mtile * 16 + (lane & 15);
  int kh  = lane >> 4;
  f32x8 acc = {0.f, 0.f, 0.f, 0.f, 0.f, 0.f, 0.f, 0.f};
#pragma unroll
  for (int kk = 0; kk < DIM2; kk += 32) {
    int kb = kk + kh * 8;
    const float* cp = C1 + (size_t)row * DIM2 + kb;
    bf16x16 a;
#pragma unroll
    for (int j = 0; j < 8; ++j)
      a[j]     = (__bf16)fmaxf(cp[j]      * scale[kb + j]      + shift[kb + j],      0.f);
#pragma unroll
    for (int j = 0; j < 8; ++j)
      a[8 + j] = (__bf16)fmaxf(cp[16 + j] * scale[kb + 16 + j] + shift[kb + 16 + j], 0.f);
    bf16x16 b = *(const bf16x16*)(Bswz + (((size_t)(kk >> 5) * NT + ntile) * 32 + lane) * 16);
    acc = __builtin_amdgcn_wmma_f32_16x16x32_bf16(false, a, false, b, (short)0, acc, false, false);
  }
  int col   = ntile * 16 + (lane & 15);
  int rbase = mtile * 16 + kh * 8;
#pragma unroll
  for (int i = 0; i < 8; ++i) C2[(size_t)(rbase + i) * DIM + col] = acc[i];
}

// ---------------------------------------------------------------------------
// per-column sum / sumsq for BN batch stats (coalesced, block-strided rows)
// ---------------------------------------------------------------------------
__global__ void gin_col_stats(const float* __restrict__ C, int ncols,
                              float* __restrict__ sum, float* __restrict__ sumsq) {
  int c = threadIdx.x; // blockDim.x == ncols
  float s = 0.f, s2 = 0.f;
  for (int r = blockIdx.x; r < NNODES; r += gridDim.x) {
    float v = C[(size_t)r * ncols + c];
    s += v; s2 += v * v;
  }
  unsafeAtomicAdd(&sum[c], s);
  unsafeAtomicAdd(&sumsq[c], s2);
}

// scale = g * rsqrt(var+eps); shift = beta - mean*scale  (bias pre-BN cancels)
__global__ void gin_bn_finalize(const float* __restrict__ sum, const float* __restrict__ sumsq,
                                const float* __restrict__ g, const float* __restrict__ be,
                                float* __restrict__ scale, float* __restrict__ shift, int ncols) {
  int c = blockIdx.x * blockDim.x + threadIdx.x;
  if (c >= ncols) return;
  const float inv_n = 1.0f / (float)NNODES;
  float mean = sum[c] * inv_n;
  float var  = sumsq[c] * inv_n - mean * mean;
  float sc   = g[c] * rsqrtf(var + BN_EPS);
  scale[c] = sc;
  shift[c] = be[c] - mean * sc;
}

// h = BN2(C2) [+ relu for non-final layer]
__global__ void gin_bn_apply(const float* __restrict__ C2, const float* __restrict__ scale,
                             const float* __restrict__ shift, float* __restrict__ hout, int relu) {
  int tid = blockIdx.x * blockDim.x + threadIdx.x; // one float4 per thread
  if (tid >= NNODES * DIM / 4) return;
  int c = (tid * 4) & (DIM - 1);
  float4 v  = *(const float4*)(C2 + (size_t)tid * 4);
  float4 sc = *(const float4*)(scale + c);
  float4 sh = *(const float4*)(shift + c);
  float4 o;
  o.x = v.x * sc.x + sh.x;
  o.y = v.y * sc.y + sh.y;
  o.z = v.z * sc.z + sh.z;
  o.w = v.w * sc.w + sh.w;
  if (relu) {
    o.x = fmaxf(o.x, 0.f); o.y = fmaxf(o.y, 0.f);
    o.z = fmaxf(o.z, 0.f); o.w = fmaxf(o.w, 0.f);
  }
  *(float4*)(hout + (size_t)tid * 4) = o;
}

extern "C" void kernel_launch(void* const* d_in, const int* in_sizes, int n_in,
                              void* d_out, int out_size, void* d_ws, size_t ws_size,
                              hipStream_t stream) {
  (void)in_sizes; (void)n_in; (void)out_size; (void)ws_size;
  const int*   x          = (const int*)d_in[0];
  const int*   z          = (const int*)d_in[1];
  const int*   edge_index = (const int*)d_in[2];
  const int*   edge_attr  = (const int*)d_in[3];
  const float* atom_emb   = (const float*)d_in[4];
  const float* z_emb      = (const float*)d_in[5];
  const float* bond_emb   = (const float*)d_in[6];
  const float* epsv       = (const float*)d_in[7];
  const float* W1         = (const float*)d_in[8];
  // d_in[9] = b1 (cancels under BN), d_in[13] = b2 (cancels under BN)
  const float* g1         = (const float*)d_in[10];
  const float* be1        = (const float*)d_in[11];
  const float* W2         = (const float*)d_in[12];
  const float* bng        = (const float*)d_in[14];
  const float* bnb        = (const float*)d_in[15];

  float* h = (float*)d_out; // h lives in d_out; fully rewritten every call

  char* ws = (char*)d_ws;
  float*  C1  = (float*)ws;                 ws += (size_t)NNODES * DIM2 * sizeof(float);
  float*  agg = (float*)ws;                 // aliased with C2 (agg dead before GEMM2 writes)
  float*  C2  = agg;                        ws += (size_t)NNODES * DIM * sizeof(float);
  __bf16* W1s = (__bf16*)ws;                ws += (size_t)NLAYERS * DIM * DIM2 * sizeof(__bf16);
  __bf16* W2s = (__bf16*)ws;                ws += (size_t)NLAYERS * DIM2 * DIM * sizeof(__bf16);
  float*  ssum  = (float*)ws;               ws += 256 * sizeof(float);
  float*  ssq   = (float*)ws;               ws += 256 * sizeof(float);
  float*  scale = (float*)ws;               ws += 256 * sizeof(float);
  float*  shift = (float*)ws;               ws += 256 * sizeof(float);

  // pre-swizzle weights into WMMA B-fragment order (tiny: 2x64K elems each)
  for (int l = 0; l < NLAYERS; ++l) {
    gin_convert_w<<<(DIM * DIM2 + 255) / 256, 256, 0, stream>>>(
        W1 + (size_t)l * DIM * DIM2, W1s + (size_t)l * DIM * DIM2, DIM, DIM2);
    gin_convert_w<<<(DIM2 * DIM + 255) / 256, 256, 0, stream>>>(
        W2 + (size_t)l * DIM2 * DIM, W2s + (size_t)l * DIM2 * DIM, DIM2, DIM);
  }

  gin_node_embed<<<NNODES / 8, 256, 0, stream>>>(x, z, atom_emb, z_emb, h);

  for (int l = 0; l < NLAYERS; ++l) {
    hipMemsetAsync(agg, 0, (size_t)NNODES * DIM * sizeof(float), stream);
    gin_edge_message<<<NEDGES / 8, 256, 0, stream>>>(
        edge_index, edge_attr, bond_emb + (size_t)l * 3 * 6 * DIM, h, agg);

    gin_gemm1<<<(NNODES / 16) * (DIM2 / 16) / 8, 256, 0, stream>>>(
        h, agg, epsv, l, W1s + (size_t)l * DIM * DIM2, C1);

    hipMemsetAsync(ssum, 0, 256 * sizeof(float), stream);
    hipMemsetAsync(ssq,  0, 256 * sizeof(float), stream);
    gin_col_stats<<<512, DIM2, 0, stream>>>(C1, DIM2, ssum, ssq);
    gin_bn_finalize<<<1, DIM2, 0, stream>>>(ssum, ssq, g1 + l * DIM2, be1 + l * DIM2,
                                            scale, shift, DIM2);

    gin_gemm2<<<(NNODES / 16) * (DIM / 16) / 8, 256, 0, stream>>>(
        C1, scale, shift, W2s + (size_t)l * DIM2 * DIM, C2);

    hipMemsetAsync(ssum, 0, 256 * sizeof(float), stream);
    hipMemsetAsync(ssq,  0, 256 * sizeof(float), stream);
    gin_col_stats<<<512, DIM, 0, stream>>>(C2, DIM, ssum, ssq);
    gin_bn_finalize<<<1, DIM, 0, stream>>>(ssum, ssq, bng + l * DIM, bnb + l * DIM,
                                           scale, shift, DIM);

    gin_bn_apply<<<(NNODES * DIM / 4 + 255) / 256, 256, 0, stream>>>(
        C2, scale, shift, h, (l < NLAYERS - 1) ? 1 : 0);
  }
}
